// TwoLayerGAT_75342316306473
// MI455X (gfx1250) — matrix operand order repaired and checked by reading, not compiled, not run on hardware
//
#include <hip/hip_runtime.h>

// ---------------------------------------------------------------------------
// TwoLayerGAT for MI455X (gfx1250): bf16 WMMA for all dense projections,
// wave32 gather/scatter kernels for the edge-softmax phase.
// ---------------------------------------------------------------------------

typedef __attribute__((ext_vector_type(16))) __bf16 v16bf;
typedef __attribute__((ext_vector_type(8)))  __bf16 v8bf;
typedef __attribute__((ext_vector_type(8)))  float  v8f;

static constexpr int N_NODES = 50000;
static constexpr int E_EDGES = 800000;
static constexpr int DIN     = 256;
static constexpr int D       = 128;   // = H*C
static constexpr int H       = 4;
static constexpr int NC      = 40;

// ---- order-preserving float <-> uint encoding for atomic segment-max ------
static __device__ __forceinline__ unsigned encf(float f) {
    unsigned u = __float_as_uint(f);
    return (u & 0x80000000u) ? ~u : (u | 0x80000000u);
}
static __device__ __forceinline__ float decf(unsigned e) {
    return (e & 0x80000000u) ? __uint_as_float(e ^ 0x80000000u)
                             : __uint_as_float(~e);
}

// ---------------------------------------------------------------------------
__global__ void fill_u32_kernel(unsigned* __restrict__ p, unsigned v, int n) {
    int i = blockIdx.x * blockDim.x + threadIdx.x;
    if (i < n) p[i] = v;
}

__global__ void cvt_bf16_kernel(const float* __restrict__ in,
                                __bf16* __restrict__ out, int n) {
    int i = blockIdx.x * blockDim.x + threadIdx.x;
    if (i < n) out[i] = (__bf16)in[i];
}

// ---------------------------------------------------------------------------
// C[M x Nout] = A[M x K] * B[Nout x K]^T   (both row-major, K-contiguous)
// One wave -> 16(M) x 64(N) tile, K stepped by 32, bf16 WMMA, f32 accum.
// M % 16 == 0, K % 32 == 0, Nout % 64 == 0 guaranteed by problem sizes.
// ---------------------------------------------------------------------------
__global__ void gemm_bf16_wmma_kernel(const __bf16* __restrict__ A,
                                      const __bf16* __restrict__ B,
                                      float* __restrict__ Cmat,
                                      int M, int K, int Nout) {
    const int lane  = threadIdx.x & 31;
    const int wave  = (blockIdx.x * blockDim.x + threadIdx.x) >> 5;
    const int nWT   = Nout >> 6;              // 64 columns per wave
    const int mTile = wave / nWT;
    const int nBase = (wave % nWT) << 6;
    if (mTile >= (M >> 4)) return;            // wave-uniform exit (EXEC all-1)
    const int half = lane >> 4;
    const int l15  = lane & 15;

    const __bf16* a_row = A + (size_t)(mTile * 16 + l15) * K;
    const __bf16* b_row0 = B + (size_t)(nBase +  0 + l15) * K;
    const __bf16* b_row1 = B + (size_t)(nBase + 16 + l15) * K;
    const __bf16* b_row2 = B + (size_t)(nBase + 32 + l15) * K;
    const __bf16* b_row3 = B + (size_t)(nBase + 48 + l15) * K;

    v8f acc0 = {}, acc1 = {}, acc2 = {}, acc3 = {};
    for (int k0 = 0; k0 < K; k0 += 32) {
        // A fragment: lane holds contiguous K = [half*16, half*16+16)
        v8bf alo = *(const v8bf*)(a_row + k0 + half * 16);
        v8bf ahi = *(const v8bf*)(a_row + k0 + half * 16 + 8);
        v16bf af;
#pragma unroll
        for (int i = 0; i < 8; ++i) { af[i] = alo[i]; af[i + 8] = ahi[i]; }
        // B fragment: lane holds contiguous K = [half*16, half*16+16)
        v16bf bf0 = *(const v16bf*)(b_row0 + k0 + half * 16);
        v16bf bf1 = *(const v16bf*)(b_row1 + k0 + half * 16);
        v16bf bf2 = *(const v16bf*)(b_row2 + k0 + half * 16);
        v16bf bf3 = *(const v16bf*)(b_row3 + k0 + half * 16);
        acc0 = __builtin_amdgcn_wmma_f32_16x16x32_bf16(false, af, false, bf0,
                                                       (short)0, acc0, false, false);
        acc1 = __builtin_amdgcn_wmma_f32_16x16x32_bf16(false, af, false, bf1,
                                                       (short)0, acc1, false, false);
        acc2 = __builtin_amdgcn_wmma_f32_16x16x32_bf16(false, af, false, bf2,
                                                       (short)0, acc2, false, false);
        acc3 = __builtin_amdgcn_wmma_f32_16x16x32_bf16(false, af, false, bf3,
                                                       (short)0, acc3, false, false);
    }
    // C/D layout: VGPR v -> row (half*8 + v), column = lane&15 within tile
    float* crow = Cmat + (size_t)(mTile * 16 + half * 8) * Nout + nBase + l15;
#pragma unroll
    for (int v = 0; v < 8; ++v) {
        crow[(size_t)v * Nout +  0] = acc0[v];
        crow[(size_t)v * Nout + 16] = acc1[v];
        crow[(size_t)v * Nout + 32] = acc2[v];
        crow[(size_t)v * Nout + 48] = acc3[v];
    }
}

// ---------------------------------------------------------------------------
// LayerNorm(row of 128) + ReLU, one wave per node; also emit bf16 copy.
// ---------------------------------------------------------------------------
__global__ void ln_relu_kernel(float* __restrict__ h,
                               const float* __restrict__ gamma,
                               const float* __restrict__ beta,
                               __bf16* __restrict__ hbf) {
    int wave = (blockIdx.x * blockDim.x + threadIdx.x) >> 5;
    int lane = threadIdx.x & 31;
    if (wave >= N_NODES) return;
    float* row = h + (size_t)wave * D;
    float4 v = *(const float4*)(row + lane * 4);
    float s = v.x + v.y + v.z + v.w;
#pragma unroll
    for (int off = 16; off; off >>= 1) s += __shfl_xor(s, off, 32);
    float mu = s * (1.0f / 128.0f);
    float dx = v.x - mu, dy = v.y - mu, dz = v.z - mu, dw = v.w - mu;
    float q = dx * dx + dy * dy + dz * dz + dw * dw;
#pragma unroll
    for (int off = 16; off; off >>= 1) q += __shfl_xor(q, off, 32);
    float inv = rsqrtf(q * (1.0f / 128.0f) + 1e-5f);
    int c = lane * 4;
    float o0 = fmaxf(dx * inv * gamma[c + 0] + beta[c + 0], 0.0f);
    float o1 = fmaxf(dy * inv * gamma[c + 1] + beta[c + 1], 0.0f);
    float o2 = fmaxf(dz * inv * gamma[c + 2] + beta[c + 2], 0.0f);
    float o3 = fmaxf(dw * inv * gamma[c + 3] + beta[c + 3], 0.0f);
    row[c + 0] = o0; row[c + 1] = o1; row[c + 2] = o2; row[c + 3] = o3;
    __bf16* brow = hbf + (size_t)wave * D + c;
    brow[0] = (__bf16)o0; brow[1] = (__bf16)o1;
    brow[2] = (__bf16)o2; brow[3] = (__bf16)o3;
}

// ---------------------------------------------------------------------------
// Per-edge attention score (wave per edge): s[e][h] = sum_c lrelu(m)*att
// plus atomic segment-max into smax_enc[dst][h].
// ---------------------------------------------------------------------------
__global__ void edge_score_kernel(const float* __restrict__ xl,
                                  const float* __restrict__ xr,
                                  const long long* __restrict__ src,
                                  const long long* __restrict__ dst,
                                  const float* __restrict__ ew,
                                  const float* __restrict__ We,
                                  const float* __restrict__ att,
                                  float* __restrict__ s_out,
                                  unsigned* __restrict__ smax_enc) {
    int e = (blockIdx.x * blockDim.x + threadIdx.x) >> 5;
    int lane = threadIdx.x & 31;
    if (e >= E_EDGES) return;
    long long si = src[e], di = dst[e];
    float w = ew[e];
    const float* xls = xl + (size_t)si * D;
    const float* xrd = xr + (size_t)di * D;
    float p[4];
#pragma unroll
    for (int hh = 0; hh < 4; ++hh) {
        int c = hh * 32 + lane;
        float m = xls[c] + xrd[c] + w * We[c];
        float lr = m > 0.0f ? m : 0.2f * m;
        p[hh] = lr * att[c];
    }
#pragma unroll
    for (int off = 16; off; off >>= 1) {
        p[0] += __shfl_xor(p[0], off, 32);
        p[1] += __shfl_xor(p[1], off, 32);
        p[2] += __shfl_xor(p[2], off, 32);
        p[3] += __shfl_xor(p[3], off, 32);
    }
    if (lane < 4) {
        float sh = (lane == 0) ? p[0] : (lane == 1) ? p[1] : (lane == 2) ? p[2] : p[3];
        s_out[(size_t)e * 4 + lane] = sh;
        atomicMax(smax_enc + (size_t)di * 4 + lane, encf(sh));
    }
}

// ---------------------------------------------------------------------------
// ex = exp(s - smax[dst]); in-place over s; atomicAdd denom. Thread per (e,h).
// ---------------------------------------------------------------------------
__global__ void edge_exp_kernel(float* __restrict__ s,
                                const long long* __restrict__ dst,
                                const unsigned* __restrict__ smax_enc,
                                float* __restrict__ denom) {
    int i = blockIdx.x * blockDim.x + threadIdx.x;
    if (i >= E_EDGES * H) return;
    int e = i >> 2, hh = i & 3;
    long long di = dst[e];
    float sm = decf(smax_enc[(size_t)di * 4 + hh]);
    if (!(sm >= -3.0e38f && sm <= 3.0e38f)) sm = 0.0f;   // jnp.where(isfinite)
    float ex = __expf(s[i] - sm);
    s[i] = ex;
    atomicAdd(denom + (size_t)di * 4 + hh, ex);
}

// ---------------------------------------------------------------------------
// acc[dst] += xl[src] * alpha   (wave per edge; 4 f32 atomics per lane)
// ---------------------------------------------------------------------------
__global__ void edge_agg_kernel(const float* __restrict__ xl,
                                const long long* __restrict__ src,
                                const long long* __restrict__ dst,
                                const float* __restrict__ ex,
                                const float* __restrict__ denom,
                                float* __restrict__ acc) {
    int e = (blockIdx.x * blockDim.x + threadIdx.x) >> 5;
    int lane = threadIdx.x & 31;
    if (e >= E_EDGES) return;
    long long si = src[e], di = dst[e];
    float a0 = ex[(size_t)e * 4 + 0] / (denom[(size_t)di * 4 + 0] + 1e-16f);
    float a1 = ex[(size_t)e * 4 + 1] / (denom[(size_t)di * 4 + 1] + 1e-16f);
    float a2 = ex[(size_t)e * 4 + 2] / (denom[(size_t)di * 4 + 2] + 1e-16f);
    float a3 = ex[(size_t)e * 4 + 3] / (denom[(size_t)di * 4 + 3] + 1e-16f);
    const float* xls = xl + (size_t)si * D;
    float* out = acc + (size_t)di * D;
    atomicAdd(out +       lane, xls[      lane] * a0);
    atomicAdd(out +  32 + lane, xls[ 32 + lane] * a1);
    atomicAdd(out +  64 + lane, xls[ 64 + lane] * a2);
    atomicAdd(out +  96 + lane, xls[ 96 + lane] * a3);
}

// ---------------------------------------------------------------------------
__global__ void finish_elu_bf16_kernel(const float* __restrict__ acc,
                                       const float* __restrict__ b,
                                       __bf16* __restrict__ out_bf, int total) {
    int i = blockIdx.x * blockDim.x + threadIdx.x;
    if (i >= total) return;
    float v = acc[i] + b[i & (D - 1)];
    v = v > 0.0f ? v : (expf(v) - 1.0f);
    out_bf[i] = (__bf16)v;
}

__global__ void finish_residual_kernel(const float* __restrict__ acc,
                                       const float* __restrict__ b,
                                       const float* __restrict__ h0,
                                       float* __restrict__ hout, int total) {
    int i = blockIdx.x * blockDim.x + threadIdx.x;
    if (i >= total) return;
    float v = acc[i] + b[i & (D - 1)];
    v = v > 0.0f ? v : (expf(v) - 1.0f);
    hout[i] = h0[i] + v;
}

// ---------------------------------------------------------------------------
__global__ void logits_kernel(const float* __restrict__ h,
                              const float* __restrict__ Wc,
                              const float* __restrict__ bc,
                              float* __restrict__ out) {
    int i = blockIdx.x * blockDim.x + threadIdx.x;
    if (i >= N_NODES * NC) return;
    int n = i / NC, c = i - n * NC;
    const float* hr = h + (size_t)n * D;
    const float* wr = Wc + (size_t)c * D;
    float s = 0.0f;
#pragma unroll 8
    for (int k = 0; k < D; ++k) s = fmaf(hr[k], wr[k], s);
    out[i] = s + bc[c];
}

// ---------------------------------------------------------------------------
extern "C" void kernel_launch(void* const* d_in, const int* in_sizes, int n_in,
                              void* d_out, int out_size, void* d_ws, size_t ws_size,
                              hipStream_t stream) {
    (void)in_sizes; (void)n_in; (void)out_size; (void)ws_size;
    const float*     x     = (const float*)d_in[0];
    const long long* eidx  = (const long long*)d_in[1];
    const float*     ew    = (const float*)d_in[2];
    const float*     W_pre = (const float*)d_in[3];
    const float*     gamma = (const float*)d_in[4];
    const float*     beta  = (const float*)d_in[5];
    const float*     Wl1   = (const float*)d_in[6];
    const float*     Wr1   = (const float*)d_in[7];
    const float*     We1   = (const float*)d_in[8];
    const float*     att1  = (const float*)d_in[9];
    const float*     b1    = (const float*)d_in[10];
    const float*     Wl2   = (const float*)d_in[11];
    const float*     Wr2   = (const float*)d_in[12];
    const float*     We2   = (const float*)d_in[13];
    const float*     att2  = (const float*)d_in[14];
    const float*     b2    = (const float*)d_in[15];
    const float*     Wc    = (const float*)d_in[16];
    const float*     bc    = (const float*)d_in[17];

    const long long* src = eidx;
    const long long* dst = eidx + E_EDGES;

    // ---- workspace carve (256 B aligned) ----
    char* ws = (char*)d_ws; size_t off = 0;
    auto carve = [&](size_t bytes) -> void* {
        void* p = ws + off; off += (bytes + 255) & ~(size_t)255; return p;
    };
    __bf16*   x_bf    = (__bf16*)carve((size_t)N_NODES * DIN * 2);
    __bf16*   wpre_bf = (__bf16*)carve((size_t)D * DIN * 2);
    __bf16*   wl1_bf  = (__bf16*)carve((size_t)D * D * 2);
    __bf16*   wr1_bf  = (__bf16*)carve((size_t)D * D * 2);
    __bf16*   wl2_bf  = (__bf16*)carve((size_t)D * D * 2);
    __bf16*   wr2_bf  = (__bf16*)carve((size_t)D * D * 2);
    float*    h0      = (float*)carve((size_t)N_NODES * D * 4);
    __bf16*   h0_bf   = (__bf16*)carve((size_t)N_NODES * D * 2);
    __bf16*   h1_bf   = (__bf16*)carve((size_t)N_NODES * D * 2);
    float*    xl      = (float*)carve((size_t)N_NODES * D * 4);
    float*    xr      = (float*)carve((size_t)N_NODES * D * 4);
    float*    accb    = (float*)carve((size_t)N_NODES * D * 4);
    float*    sbuf    = (float*)carve((size_t)E_EDGES * H * 4);
    unsigned* smax    = (unsigned*)carve((size_t)N_NODES * H * 4);
    float*    denom   = (float*)carve((size_t)N_NODES * H * 4);

    const int TB = 256;
    auto cdiv = [](long long a, long long b) { return (int)((a + b - 1) / b); };
    const unsigned ENC_NEG_INF = 0x007FFFFFu;   // encf(-inf)

    // ---- bf16 conversions ----
    cvt_bf16_kernel<<<cdiv((long long)N_NODES * DIN, TB), TB, 0, stream>>>(x, x_bf, N_NODES * DIN);
    cvt_bf16_kernel<<<cdiv(D * DIN, TB), TB, 0, stream>>>(W_pre, wpre_bf, D * DIN);
    cvt_bf16_kernel<<<cdiv(D * D, TB), TB, 0, stream>>>(Wl1, wl1_bf, D * D);
    cvt_bf16_kernel<<<cdiv(D * D, TB), TB, 0, stream>>>(Wr1, wr1_bf, D * D);
    cvt_bf16_kernel<<<cdiv(D * D, TB), TB, 0, stream>>>(Wl2, wl2_bf, D * D);
    cvt_bf16_kernel<<<cdiv(D * D, TB), TB, 0, stream>>>(Wr2, wr2_bf, D * D);

    const int gemmWaves  = (N_NODES / 16) * (D / 64);     // 6250 waves
    const int gemmBlocks = cdiv(gemmWaves, TB / 32);

    // ---- pre projection + LN + ReLU ----
    gemm_bf16_wmma_kernel<<<gemmBlocks, TB, 0, stream>>>(x_bf, wpre_bf, h0, N_NODES, DIN, D);
    ln_relu_kernel<<<cdiv((long long)N_NODES * 32, TB), TB, 0, stream>>>(h0, gamma, beta, h0_bf);

    // ---- GAT layer 1 ----
    gemm_bf16_wmma_kernel<<<gemmBlocks, TB, 0, stream>>>(h0_bf, wl1_bf, xl, N_NODES, D, D);
    gemm_bf16_wmma_kernel<<<gemmBlocks, TB, 0, stream>>>(h0_bf, wr1_bf, xr, N_NODES, D, D);
    fill_u32_kernel<<<cdiv(N_NODES * H, TB), TB, 0, stream>>>(smax, ENC_NEG_INF, N_NODES * H);
    fill_u32_kernel<<<cdiv(N_NODES * H, TB), TB, 0, stream>>>((unsigned*)denom, 0u, N_NODES * H);
    fill_u32_kernel<<<cdiv((long long)N_NODES * D, TB), TB, 0, stream>>>((unsigned*)accb, 0u, N_NODES * D);
    edge_score_kernel<<<cdiv((long long)E_EDGES * 32, TB), TB, 0, stream>>>(xl, xr, src, dst, ew, We1, att1, sbuf, smax);
    edge_exp_kernel<<<cdiv((long long)E_EDGES * H, TB), TB, 0, stream>>>(sbuf, dst, smax, denom);
    edge_agg_kernel<<<cdiv((long long)E_EDGES * 32, TB), TB, 0, stream>>>(xl, src, dst, sbuf, denom, accb);
    finish_elu_bf16_kernel<<<cdiv((long long)N_NODES * D, TB), TB, 0, stream>>>(accb, b1, h1_bf, N_NODES * D);

    // ---- GAT layer 2 ----
    gemm_bf16_wmma_kernel<<<gemmBlocks, TB, 0, stream>>>(h1_bf, wl2_bf, xl, N_NODES, D, D);
    gemm_bf16_wmma_kernel<<<gemmBlocks, TB, 0, stream>>>(h1_bf, wr2_bf, xr, N_NODES, D, D);
    fill_u32_kernel<<<cdiv(N_NODES * H, TB), TB, 0, stream>>>(smax, ENC_NEG_INF, N_NODES * H);
    fill_u32_kernel<<<cdiv(N_NODES * H, TB), TB, 0, stream>>>((unsigned*)denom, 0u, N_NODES * H);
    fill_u32_kernel<<<cdiv((long long)N_NODES * D, TB), TB, 0, stream>>>((unsigned*)accb, 0u, N_NODES * D);
    edge_score_kernel<<<cdiv((long long)E_EDGES * 32, TB), TB, 0, stream>>>(xl, xr, src, dst, ew, We2, att2, sbuf, smax);
    edge_exp_kernel<<<cdiv((long long)E_EDGES * H, TB), TB, 0, stream>>>(sbuf, dst, smax, denom);
    edge_agg_kernel<<<cdiv((long long)E_EDGES * 32, TB), TB, 0, stream>>>(xl, src, dst, sbuf, denom, accb);

    // ---- residual + outputs ----
    float* h_final = (float*)d_out;
    float* logits  = (float*)d_out + (size_t)N_NODES * D;
    finish_residual_kernel<<<cdiv((long long)N_NODES * D, TB), TB, 0, stream>>>(accb, b2, h0, h_final, N_NODES * D);
    logits_kernel<<<cdiv((long long)N_NODES * NC, TB), TB, 0, stream>>>(h_final, Wc, bc, logits);
}
